// TorchEpsilonSvr_65137474011866
// MI455X (gfx1250) — matrix operand order repaired and checked by reading, not compiled, not run on hardware
//
#include <hip/hip_runtime.h>
#include <hip/hip_bf16.h>

typedef __attribute__((ext_vector_type(16))) _Float16 v16h;
typedef __attribute__((ext_vector_type(8)))  _Float16 v8h;
typedef __attribute__((ext_vector_type(8)))  float    v8f;
typedef int b128i __attribute__((vector_size(16)));   // payload type for async b128

#define GAMMA 0.01f
// K2 = -gamma * log2(e);  exp(-gamma*d2) == exp2(K2*d2)
#define K2   (-0.014426950408889634f)
#define M2K2 (0.028853900817779268f)   // -2*K2

#if defined(__AMDGCN__) && __has_builtin(__builtin_amdgcn_global_load_async_to_lds_b128)
#define USE_ASYNC_COPY 1
#else
#define USE_ASYNC_COPY 0
#endif

__device__ __forceinline__ v8h cvt8(float4 a, float4 b) {
    v8h h;
    h[0] = (_Float16)a.x; h[1] = (_Float16)a.y; h[2] = (_Float16)a.z; h[3] = (_Float16)a.w;
    h[4] = (_Float16)b.x; h[5] = (_Float16)b.y; h[6] = (_Float16)b.z; h[7] = (_Float16)b.w;
    return h;
}
__device__ __forceinline__ float sq4(float4 a) {
    return a.x * a.x + a.y * a.y + a.z * a.z + a.w * a.w;
}
__device__ __forceinline__ float fast_exp2(float x) {
#if defined(__AMDGCN__) && __has_builtin(__builtin_amdgcn_exp2f)
    return __builtin_amdgcn_exp2f(x);
#else
    return exp2f(x);
#endif
}
// 16B global -> LDS stage: async DMA (ASYNCcnt) on CDNA5, sync copy otherwise.
__device__ __forceinline__ void stage16(const _Float16* gp, _Float16* lp) {
#if USE_ASYNC_COPY
    __builtin_amdgcn_global_load_async_to_lds_b128(
        (__attribute__((address_space(1))) b128i*)gp,
        (__attribute__((address_space(3))) b128i*)lp,
        0, 0);
#else
    *(v8h*)lp = *(const v8h*)gp;
#endif
}
__device__ __forceinline__ void wait_async0() {
#if USE_ASYNC_COPY
#if __has_builtin(__builtin_amdgcn_s_wait_asynccnt)
    __builtin_amdgcn_s_wait_asynccnt(0);
#else
    asm volatile("s_wait_asynccnt 0x0" ::: "memory");
#endif
#endif
}

// ---- Prepass: sv (f32, M x 64) -> f16 copy + K2-scaled squared row norms ----
__global__ void __launch_bounds__(256)
svr_prepass_kernel(const float* __restrict__ sv, _Float16* __restrict__ svh_g,
                   float* __restrict__ svsq_g, int M) {
    const int gid = blockIdx.x * 256 + threadIdx.x;  // one thread per 8 elems
    const int row = gid >> 3;
    const int cg  = (gid & 7) * 8;
    if (row < M) {
        const float* p = sv + (size_t)row * 64 + cg;
        float4 q0 = ((const float4*)p)[0];
        float4 q1 = ((const float4*)p)[1];
        *(v8h*)(svh_g + (size_t)row * 64 + cg) = cvt8(q0, q1);
        float s = sq4(q0) + sq4(q1);
        s += __shfl_xor(s, 1);
        s += __shfl_xor(s, 2);
        s += __shfl_xor(s, 4);
        if ((gid & 7) == 0) svsq_g[row] = K2 * s;    // pre-scaled
    }
}

// ---- Main: 64 rows of x per WG, sweep M in 32-col tiles, double-buffered
// async tile copies; 8 waves = 4 row-tiles x 2 col-tiles of 16x16 WMMA. ----
__global__ void __launch_bounds__(256)
TorchEpsilonSvr_65137474011866_kernel(const float* __restrict__ x,
                                      const _Float16* __restrict__ svh_g,
                                      const float* __restrict__ svsq_g,
                                      const float* __restrict__ coef,
                                      const float* __restrict__ intercept,
                                      float* __restrict__ out,
                                      int M) {
    __shared__ __align__(16) _Float16 xh[64][64];      // x block, f16
    __shared__ __align__(16) _Float16 svh[2][32][64];  // sv tile ping-pong
    __shared__ float xsq_s[64];
    __shared__ float partial_s[2][64];

    const int t    = threadIdx.x;
    const int lane = t & 31;
    const int w    = t >> 5;
    const int r    = w >> 1;   // row tile 0..3
    const int cid  = w & 1;    // col tile 0..1
    const size_t row0 = (size_t)blockIdx.x * 64;

    const int srow = t >> 3;        // 0..31: this thread's 16B slot of a tile
    const int scg  = (t & 7) * 8;

    // ---- Prologue: kick off tile 0's async copy before anything else ----
    stage16(svh_g + t * 8, &svh[0][srow][scg]);

    // ---- Phase 1: stage x block to LDS (f16) + row squared norms (f32) ----
    {
        const int row = t >> 2;          // 0..63
        const int cg  = (t & 3) * 16;    // 0,16,32,48
        const float* p = x + (row0 + row) * 64 + cg;
        float4 q0 = ((const float4*)p)[0];
        float4 q1 = ((const float4*)p)[1];
        float4 q2 = ((const float4*)p)[2];
        float4 q3 = ((const float4*)p)[3];
        *(v8h*)&xh[row][cg]     = cvt8(q0, q1);
        *(v8h*)&xh[row][cg + 8] = cvt8(q2, q3);
        float s = sq4(q0) + sq4(q1) + sq4(q2) + sq4(q3);
        s += __shfl_xor(s, 1);
        s += __shfl_xor(s, 2);
        if ((t & 3) == 0) xsq_s[row] = s;
    }
    __syncthreads();

    // A-operand (ISA 7.12.2 16-bit A 16x32): M = lane%16; lanes 0-15 hold
    // K=[0..7]|[16..23] in v0-3|v4-7, lanes 16-31 hold K=[8..15]|[24..31].
    const int arow = 16 * r + (lane & 15);
    const int hoff = (lane < 16) ? 0 : 8;
    v16h a0, a1;
    {
        v8h lo = *(const v8h*)&xh[arow][0 + hoff];
        v8h hi = *(const v8h*)&xh[arow][16 + hoff];
        v8h l1 = *(const v8h*)&xh[arow][32 + hoff];
        v8h h1 = *(const v8h*)&xh[arow][48 + hoff];
#pragma unroll
        for (int i = 0; i < 8; ++i) {
            a0[i] = lo[i]; a0[8 + i] = hi[i];
            a1[i] = l1[i]; a1[8 + i] = h1[i];
        }
    }
    // K2-scaled ||x||^2 for the 8 output rows this lane holds (C layout:
    // lanes 0-15 -> M = v, lanes 16-31 -> M = 8+v).
    float axv[8];
#pragma unroll
    for (int v = 0; v < 8; ++v) axv[v] = K2 * xsq_s[16 * r + hoff + v];

    const int nl = 16 * cid + (lane & 15);  // output column within tile
    float acc[8];
#pragma unroll
    for (int v = 0; v < 8; ++v) acc[v] = 0.0f;

    const int brow = 16 * cid + (lane & 15);
    const int kb   = (lane < 16) ? 0 : 8;

    const int ntiles = M >> 5;
    for (int tile = 0; tile < ntiles; ++tile) {
        const int cur = tile & 1;
        // My copy of buffer 'cur' complete; barrier makes everyone's copy
        // visible AND proves all waves finished reading buffer cur^1.
        wait_async0();
        __syncthreads();
        // Overlap: issue next tile's copy into the other buffer now.
        if (tile + 1 < ntiles)
            stage16(svh_g + (size_t)(tile + 1) * 2048 + t * 8,
                    &svh[cur ^ 1][srow][scg]);

        // B operand: N = lane%16; lanes 0-15 hold K=[0..7]|[16..23], lanes 16-31 hold
        // K=[8..15]|[24..31] (2 K-values per VGPR). Contiguous 16B LDS chunks.
        v16h b0, b1;
        {
            v8h p0 = *(const v8h*)&svh[cur][brow][kb];
            v8h p1 = *(const v8h*)&svh[cur][brow][kb + 16];
            v8h p2 = *(const v8h*)&svh[cur][brow][32 + kb];
            v8h p3 = *(const v8h*)&svh[cur][brow][32 + kb + 16];
#pragma unroll
            for (int i = 0; i < 8; ++i) {
                b0[i] = p0[i]; b0[8 + i] = p1[i];
                b1[i] = p2[i]; b1[8 + i] = p3[i];
            }
        }

        // ---- cross = x . sv^T for this 16x16 tile, f32 accumulate ----
        v8f cf = {};
        cf = __builtin_amdgcn_wmma_f32_16x16x32_f16(false, a0, false, b0,
                                                    (short)0, cf, false, false);
        cf = __builtin_amdgcn_wmma_f32_16x16x32_f16(false, a1, false, b1,
                                                    (short)0, cf, false, false);

        // ---- RBF map, fully folded: k = exp2(axv + an + (-2*K2)*cross) ----
        const int   mg  = (tile << 5) + nl;
        const float an  = svsq_g[mg];   // already K2-scaled; L2-resident
        const float cwt = coef[mg];
#pragma unroll
        for (int v = 0; v < 8; ++v) {
            const float arg = __fmaf_rn(cf[v], M2K2, axv[v] + an);
            acc[v] = __fmaf_rn(cwt, fast_exp2(arg), acc[v]);
        }
    }

    // ---- reduce over the 16 lanes that share each output row ----
#pragma unroll
    for (int v = 0; v < 8; ++v) {
        float s = acc[v];
        s += __shfl_xor(s, 1);
        s += __shfl_xor(s, 2);
        s += __shfl_xor(s, 4);
        s += __shfl_xor(s, 8);
        acc[v] = s;
    }
    if (lane == 0 || lane == 16) {
        const int h2 = (lane == 0) ? 0 : 8;
#pragma unroll
        for (int v = 0; v < 8; ++v) partial_s[cid][16 * r + h2 + v] = acc[v];
    }
    __syncthreads();
    if (t < 64) out[row0 + t] = partial_s[0][t] + partial_s[1][t] + intercept[0];
}

// ---- Fallback (no workspace): all-in-one kernel ----
__global__ void __launch_bounds__(256)
svr_fused_fallback_kernel(const float* __restrict__ x,
                          const float* __restrict__ sv,
                          const float* __restrict__ coef,
                          const float* __restrict__ intercept,
                          float* __restrict__ out,
                          int M) {
    __shared__ __align__(16) _Float16 xh[64][64];
    __shared__ __align__(16) _Float16 svh[32][64];
    __shared__ float xsq_s[64];
    __shared__ float svsq_s[32];
    __shared__ float coefs_s[32];
    __shared__ float partial_s[2][64];

    const int t    = threadIdx.x;
    const int lane = t & 31;
    const int w    = t >> 5;
    const int r    = w >> 1;
    const int cid  = w & 1;
    const size_t row0 = (size_t)blockIdx.x * 64;

    {
        const int row = t >> 2;
        const int cg  = (t & 3) * 16;
        const float* p = x + (row0 + row) * 64 + cg;
        float4 q0 = ((const float4*)p)[0];
        float4 q1 = ((const float4*)p)[1];
        float4 q2 = ((const float4*)p)[2];
        float4 q3 = ((const float4*)p)[3];
        *(v8h*)&xh[row][cg]     = cvt8(q0, q1);
        *(v8h*)&xh[row][cg + 8] = cvt8(q2, q3);
        float s = sq4(q0) + sq4(q1) + sq4(q2) + sq4(q3);
        s += __shfl_xor(s, 1);
        s += __shfl_xor(s, 2);
        if ((t & 3) == 0) xsq_s[row] = s;
    }
    __syncthreads();

    const int arow = 16 * r + (lane & 15);
    const int hoff = (lane < 16) ? 0 : 8;
    v16h a0, a1;
    {
        v8h lo = *(const v8h*)&xh[arow][0 + hoff];
        v8h hi = *(const v8h*)&xh[arow][16 + hoff];
        v8h l1 = *(const v8h*)&xh[arow][32 + hoff];
        v8h h1 = *(const v8h*)&xh[arow][48 + hoff];
#pragma unroll
        for (int i = 0; i < 8; ++i) {
            a0[i] = lo[i]; a0[8 + i] = hi[i];
            a1[i] = l1[i]; a1[8 + i] = h1[i];
        }
    }
    float axv[8];
#pragma unroll
    for (int v = 0; v < 8; ++v) axv[v] = K2 * xsq_s[16 * r + hoff + v];

    const int nl = 16 * cid + (lane & 15);
    float acc[8];
#pragma unroll
    for (int v = 0; v < 8; ++v) acc[v] = 0.0f;

    const int ntiles = M >> 5;
    for (int tile = 0; tile < ntiles; ++tile) {
        const int m0 = tile << 5;
        __syncthreads();
        {
            const int srow = t >> 3;
            const int cg   = (t & 7) * 8;
            const float* p = sv + (size_t)(m0 + srow) * 64 + cg;
            float4 q0 = ((const float4*)p)[0];
            float4 q1 = ((const float4*)p)[1];
            *(v8h*)&svh[srow][cg] = cvt8(q0, q1);
            float s = sq4(q0) + sq4(q1);
            s += __shfl_xor(s, 1);
            s += __shfl_xor(s, 2);
            s += __shfl_xor(s, 4);
            if ((t & 7) == 0) svsq_s[srow] = K2 * s;
            if (t < 32) coefs_s[t] = coef[m0 + t];
            if (tile + 1 < ntiles) __builtin_prefetch(p + 32 * 64, 0, 1);
        }
        __syncthreads();

        v16h b0, b1;
        {
            const int brow = 16 * cid + (lane & 15);
            const int kb   = (lane < 16) ? 0 : 8;
            v8h p0 = *(const v8h*)&svh[brow][kb];
            v8h p1 = *(const v8h*)&svh[brow][kb + 16];
            v8h p2 = *(const v8h*)&svh[brow][32 + kb];
            v8h p3 = *(const v8h*)&svh[brow][32 + kb + 16];
#pragma unroll
            for (int i = 0; i < 8; ++i) {
                b0[i] = p0[i]; b0[8 + i] = p1[i];
                b1[i] = p2[i]; b1[8 + i] = p3[i];
            }
        }

        v8f cf = {};
        cf = __builtin_amdgcn_wmma_f32_16x16x32_f16(false, a0, false, b0,
                                                    (short)0, cf, false, false);
        cf = __builtin_amdgcn_wmma_f32_16x16x32_f16(false, a1, false, b1,
                                                    (short)0, cf, false, false);

        const float an  = svsq_s[nl];
        const float cwt = coefs_s[nl];
#pragma unroll
        for (int v = 0; v < 8; ++v) {
            const float arg = __fmaf_rn(cf[v], M2K2, axv[v] + an);
            acc[v] = __fmaf_rn(cwt, fast_exp2(arg), acc[v]);
        }
    }

#pragma unroll
    for (int v = 0; v < 8; ++v) {
        float s = acc[v];
        s += __shfl_xor(s, 1);
        s += __shfl_xor(s, 2);
        s += __shfl_xor(s, 4);
        s += __shfl_xor(s, 8);
        acc[v] = s;
    }
    if (lane == 0 || lane == 16) {
        const int h2 = (lane == 0) ? 0 : 8;
#pragma unroll
        for (int v = 0; v < 8; ++v) partial_s[cid][16 * r + h2 + v] = acc[v];
    }
    __syncthreads();
    if (t < 64) out[row0 + t] = partial_s[0][t] + partial_s[1][t] + intercept[0];
}

extern "C" void kernel_launch(void* const* d_in, const int* in_sizes, int n_in,
                              void* d_out, int out_size, void* d_ws, size_t ws_size,
                              hipStream_t stream) {
    const float* x         = (const float*)d_in[0];
    const float* sv        = (const float*)d_in[1];
    const float* coef      = (const float*)d_in[2];
    const float* intercept = (const float*)d_in[3];
    float* out             = (float*)d_out;

    const int D = 64;
    const int N = in_sizes[0] / D;   // 16384
    const int M = in_sizes[1] / D;   // 8192

    const size_t svh_bytes = (size_t)M * D * sizeof(_Float16);
    const size_t need      = svh_bytes + (size_t)M * sizeof(float);

    if (d_ws != nullptr && ws_size >= need) {
        _Float16* svh_g = (_Float16*)d_ws;
        float*    svsq_g = (float*)((char*)d_ws + svh_bytes);
        svr_prepass_kernel<<<(M * 8 + 255) / 256, 256, 0, stream>>>(sv, svh_g, svsq_g, M);
        TorchEpsilonSvr_65137474011866_kernel<<<N / 64, 256, 0, stream>>>(
            x, svh_g, svsq_g, coef, intercept, out, M);
    } else {
        svr_fused_fallback_kernel<<<N / 64, 256, 0, stream>>>(
            x, sv, coef, intercept, out, M);
    }
}
